// MoELayerMxfp4_40570261078250
// MI455X (gfx1250) — matrix operand order repaired and checked by reading, not compiled, not run on hardware
//
#include <hip/hip_runtime.h>
#include <hip/hip_bf16.h>
#include <stdint.h>

#define T_DIM 1024
#define H_DIM 2048
#define I_DIM 4096
#define E_NUM 8
#define SLOT_BLK 8                 // slot-tiles (of 16) per wave: B dequant amortized 8x
#define MAX_SLOTS 3072             // 2048 routed slots + up to 127 pad per expert, 128-aligned

typedef __attribute__((ext_vector_type(16))) __bf16 v16bf;
typedef __attribute__((ext_vector_type(8)))  float  v8f;

union AU { v16bf v; uint4 q[2]; };
union BU { v16bf v; unsigned int u[8]; };

// ---- mxfp4(e2m1) x e8m0 -> packed 2x bf16, v_perm table decode ----
// tbl16[v] = (e<<7)|(mant<<6) (exponent field holds only e); scale fused by adding
// (s-1)<<7 per nonzero half:  e>0: exp=e+s-1 ;  e==0,m==1 (0.5): exp=s-1.  v==0 -> +-0.
__device__ __forceinline__ unsigned int fp4x2(unsigned int w, unsigned int s1) {
  const unsigned int d   = ((w >> 4) << 16) | (w & 15u);      // halfword lanes = nibbles
  const unsigned int dv  = d & 0x00070007u;                   // magnitude (perm codes 0..7)
  const unsigned int tbl = __builtin_amdgcn_perm(0x01010101u, 0x00000000u, dv << 8)   // hi bytes
                         | __builtin_amdgcn_perm(0xC0804000u, 0xC0800000u, dv);       // lo bytes
  const unsigned int nzt = (dv | (dv >> 1) | (dv >> 2)) & 0x00010001u;                // per-half !=0
  const unsigned int out = tbl + ((nzt * s1) << 7);           // add e8m0 exponent
  return out | ((d & 0x00080008u) << 12);                     // sign -> bit15
}

// 16 consecutive fp4 values (8 int32 containers, low nibble first) -> v16bf
__device__ __forceinline__ v16bf dequant16(const int* __restrict__ qp, int scale) {
  const int4 qa = *(const int4*)qp;
  const int4 qb = *(const int4*)(qp + 4);
  const unsigned int s1  = (unsigned int)(scale - 1);
  const unsigned int msk = scale ? 0xFFu : 0u;                // scale==0 -> force +0
  BU b;
  b.u[0] = fp4x2((unsigned int)qa.x & msk, s1);
  b.u[1] = fp4x2((unsigned int)qa.y & msk, s1);
  b.u[2] = fp4x2((unsigned int)qa.z & msk, s1);
  b.u[3] = fp4x2((unsigned int)qa.w & msk, s1);
  b.u[4] = fp4x2((unsigned int)qb.x & msk, s1);
  b.u[5] = fp4x2((unsigned int)qb.y & msk, s1);
  b.u[6] = fp4x2((unsigned int)qb.z & msk, s1);
  b.u[7] = fp4x2((unsigned int)qb.w & msk, s1);
  return b.v;
}

// ---------------------------------------------------------------- init
__global__ __launch_bounds__(256) void init_kernel(int* counts, int* cursor,
                                                   int* slot_token, float* slot_weight,
                                                   float* out_accum) {
  int idx = blockIdx.x * 256 + threadIdx.x;
  if (idx < E_NUM) { counts[idx] = 0; cursor[idx] = 0; }
  if (idx < MAX_SLOTS) { slot_token[idx] = -1; slot_weight[idx] = 0.f; }
  if (idx < T_DIM * H_DIM) out_accum[idx] = 0.f;
}

// ---------------------------------------------------------------- gating (one wave32 per token)
__global__ __launch_bounds__(256) void gate_kernel(const __hip_bfloat16* __restrict__ X,
                                                   const float* __restrict__ G,
                                                   int* counts, int* top2e, float* top2w) {
  const int t    = blockIdx.x * 8 + (threadIdx.x >> 5);
  const int lane = threadIdx.x & 31;
  if (t >= T_DIM) return;
  float acc[E_NUM];
#pragma unroll
  for (int e = 0; e < E_NUM; ++e) acc[e] = 0.f;
  const __hip_bfloat16* xr = X + (size_t)t * H_DIM;
  for (int h0 = lane * 4; h0 < H_DIM; h0 += 128) {
    const __hip_bfloat16* xp = xr + h0;
    float xf0 = __bfloat162float(xp[0]);
    float xf1 = __bfloat162float(xp[1]);
    float xf2 = __bfloat162float(xp[2]);
    float xf3 = __bfloat162float(xp[3]);
#pragma unroll
    for (int e = 0; e < E_NUM; ++e) {
      const float4 g4 = *(const float4*)(G + e * H_DIM + h0);
      acc[e] += xf0 * g4.x + xf1 * g4.y + xf2 * g4.z + xf3 * g4.w;
    }
  }
#pragma unroll
  for (int e = 0; e < E_NUM; ++e)
    for (int off = 16; off > 0; off >>= 1)
      acc[e] += __shfl_down(acc[e], off, 32);
  if (lane == 0) {
    float mx = acc[0];
#pragma unroll
    for (int e = 1; e < E_NUM; ++e) mx = fmaxf(mx, acc[e]);
    float p[E_NUM];
#pragma unroll
    for (int e = 0; e < E_NUM; ++e) p[e] = __expf(acc[e] - mx);
    int e0 = 0;
#pragma unroll
    for (int e = 1; e < E_NUM; ++e) if (p[e] > p[e0]) e0 = e;
    int e1 = (e0 == 0) ? 1 : 0;
#pragma unroll
    for (int e = 0; e < E_NUM; ++e) if (e != e0 && p[e] > p[e1]) e1 = e;
    float w0 = p[e0], w1 = p[e1];
    const float ws = w0 + w1;
    w0 /= ws; w1 /= ws;
    top2e[2 * t] = e0;   top2e[2 * t + 1] = e1;
    top2w[2 * t] = w0;   top2w[2 * t + 1] = w1;
    atomicAdd(&counts[e0], 1);
    atomicAdd(&counts[e1], 1);
  }
}

// ---------------------------------------------------------------- 128-padded exclusive scan
__global__ void scan_kernel(const int* counts, int* base) {
  base[0] = 0;
  for (int e = 0; e < E_NUM; ++e)
    base[e + 1] = base[e] + ((counts[e] + 127) & ~127);
}

// ---------------------------------------------------------------- token compaction
__global__ __launch_bounds__(256) void scatter_kernel(const int* top2e, const float* top2w,
                                                      const int* base, int* cursor,
                                                      int* slot_token, float* slot_weight) {
  int t = blockIdx.x * 256 + threadIdx.x;
  if (t >= T_DIM) return;
#pragma unroll
  for (int j = 0; j < 2; ++j) {
    int e = top2e[2 * t + j];
    int pos = atomicAdd(&cursor[e], 1);
    int slot = base[e] + pos;
    slot_token[slot]  = t;
    slot_weight[slot] = top2w[2 * t + j];
  }
}

// ---------------------------------------------------------------- FFN1: g = silu(X@W1^T+b1)*(X@W3^T+b3)
__global__ __launch_bounds__(256) void ffn1_kernel(const __hip_bfloat16* __restrict__ X,
                                                   const int* __restrict__ w1,
                                                   const int* __restrict__ w3,
                                                   const int* __restrict__ w13_scale,
                                                   const __hip_bfloat16* __restrict__ w13_bias,
                                                   const int* __restrict__ slot_token,
                                                   const int* __restrict__ base,
                                                   __hip_bfloat16* __restrict__ g_ws) {
  const int lane  = threadIdx.x & 31;
  const int wave  = threadIdx.x >> 5;
  const int slot0 = blockIdx.y * (16 * SLOT_BLK);
  const int i0    = blockIdx.x * 128 + wave * 16;

  int e = 0;
#pragma unroll
  for (int k = 1; k < E_NUM; ++k) if (slot0 >= base[k]) e = k;

  const int m  = lane & 15;
  const int hi = lane >> 4;

  int tok[SLOT_BLK];
  bool any = false;
#pragma unroll
  for (int j = 0; j < SLOT_BLK; ++j) {
    tok[j] = slot_token[slot0 + j * 16 + m];
    any |= (tok[j] >= 0);
  }
  if (__ballot(any) == 0) return;   // fully padded slot block

  const __hip_bfloat16* xrow[SLOT_BLK];
#pragma unroll
  for (int j = 0; j < SLOT_BLK; ++j)
    xrow[j] = X + (size_t)(tok[j] >= 0 ? tok[j] : 0) * H_DIM;   // junk rows guarded at combine

  const int koffA  = hi ? 16 : 0;
  const int khalfB = hi ? 16 : 0;
  const size_t wrow  = ((size_t)e * I_DIM + (i0 + m)) * (H_DIM / 2);
  const size_t s1row = ((size_t)e * 2 * I_DIM + (i0 + m)) * (H_DIM / 32);
  const size_t s3row = ((size_t)e * 2 * I_DIM + I_DIM + (i0 + m)) * (H_DIM / 32);

  v8f acc1[SLOT_BLK] = {};
  v8f acc3[SLOT_BLK] = {};

  for (int kk = 0; kk < H_DIM; kk += 32) {
    const int s1v = w13_scale[s1row + (kk >> 5)];
    const int s3v = w13_scale[s3row + (kk >> 5)];
    const v16bf b1 = dequant16(w1 + wrow + ((kk + khalfB) >> 1), s1v);
    const v16bf b3 = dequant16(w3 + wrow + ((kk + khalfB) >> 1), s3v);
#pragma unroll
    for (int j = 0; j < SLOT_BLK; ++j) {
      AU a;
      a.q[0] = *(const uint4*)(xrow[j] + kk + koffA);
      a.q[1] = *(const uint4*)(xrow[j] + kk + 8 + koffA);
      acc1[j] = __builtin_amdgcn_wmma_f32_16x16x32_bf16(false, a.v, false, b1, (short)0, acc1[j], false, false);
      acc3[j] = __builtin_amdgcn_wmma_f32_16x16x32_bf16(false, a.v, false, b3, (short)0, acc3[j], false, false);
    }
  }

  const int i = i0 + m;
  const float b1f = __bfloat162float(w13_bias[(size_t)e * 2 * I_DIM + i]);
  const float b3f = __bfloat162float(w13_bias[(size_t)e * 2 * I_DIM + I_DIM + i]);
#pragma unroll
  for (int j = 0; j < SLOT_BLK; ++j) {
#pragma unroll
    for (int r = 0; r < 8; ++r) {
      const int M = r + (hi ? 8 : 0);
      const float h1 = acc1[j][r] + b1f;
      const float h3 = acc3[j][r] + b3f;
      const float gv = (h1 / (1.f + __expf(-h1))) * h3;   // silu(h1)*h3
      g_ws[(size_t)(slot0 + j * 16 + M) * I_DIM + i] = __float2bfloat16(gv);
    }
  }
}

// ---------------------------------------------------------------- FFN2: out += w * (g@W2^T + b2)
__global__ __launch_bounds__(256) void ffn2_kernel(const __hip_bfloat16* __restrict__ g_ws,
                                                   const int* __restrict__ w2,
                                                   const int* __restrict__ w2_scale,
                                                   const __hip_bfloat16* __restrict__ w2_bias,
                                                   const int* __restrict__ slot_token,
                                                   const float* __restrict__ slot_weight,
                                                   const int* __restrict__ base,
                                                   float* __restrict__ out_accum) {
  const int lane  = threadIdx.x & 31;
  const int wave  = threadIdx.x >> 5;
  const int slot0 = blockIdx.y * (16 * SLOT_BLK);
  const int h0    = blockIdx.x * 128 + wave * 16;

  int e = 0;
#pragma unroll
  for (int k = 1; k < E_NUM; ++k) if (slot0 >= base[k]) e = k;

  const int m  = lane & 15;
  const int hi = lane >> 4;

  int tok[SLOT_BLK];
  bool any = false;
#pragma unroll
  for (int j = 0; j < SLOT_BLK; ++j) {
    tok[j] = slot_token[slot0 + j * 16 + m];
    any |= (tok[j] >= 0);
  }
  if (__ballot(any) == 0) return;

  const int koffA  = hi ? 16 : 0;
  const int khalfB = hi ? 16 : 0;
  const size_t w2row = ((size_t)e * H_DIM + (h0 + m)) * (I_DIM / 2);
  const size_t srow  = ((size_t)e * H_DIM + (h0 + m)) * (I_DIM / 32);
  const __hip_bfloat16* grow[SLOT_BLK];
#pragma unroll
  for (int j = 0; j < SLOT_BLK; ++j)
    grow[j] = g_ws + (size_t)(slot0 + j * 16 + m) * I_DIM;

  v8f acc[SLOT_BLK] = {};

  for (int kk = 0; kk < I_DIM; kk += 32) {
    const int sv = w2_scale[srow + (kk >> 5)];
    const v16bf b = dequant16(w2 + w2row + ((kk + khalfB) >> 1), sv);
#pragma unroll
    for (int j = 0; j < SLOT_BLK; ++j) {
      AU a;
      a.q[0] = *(const uint4*)(grow[j] + kk + koffA);
      a.q[1] = *(const uint4*)(grow[j] + kk + 8 + koffA);
      acc[j] = __builtin_amdgcn_wmma_f32_16x16x32_bf16(false, a.v, false, b, (short)0, acc[j], false, false);
    }
  }

  const int h = h0 + m;
  const float bias = __bfloat162float(w2_bias[(size_t)e * H_DIM + h]);
#pragma unroll
  for (int j = 0; j < SLOT_BLK; ++j) {
#pragma unroll
    for (int r = 0; r < 8; ++r) {
      const int M = r + (hi ? 8 : 0);
      const int slot = slot0 + j * 16 + M;
      const int t = slot_token[slot];
      if (t >= 0) {
        const float w = slot_weight[slot];
        atomicAdd(&out_accum[(size_t)t * H_DIM + h], w * (acc[j][r] + bias));
      }
    }
  }
}

// ---------------------------------------------------------------- f32 -> bf16
__global__ __launch_bounds__(256) void finalize_kernel(const float* __restrict__ out_accum,
                                                       __hip_bfloat16* __restrict__ out) {
  int idx = blockIdx.x * 256 + threadIdx.x;
  if (idx < T_DIM * H_DIM) out[idx] = __float2bfloat16(out_accum[idx]);
}

// ---------------------------------------------------------------- launch
extern "C" void kernel_launch(void* const* d_in, const int* in_sizes, int n_in,
                              void* d_out, int out_size, void* d_ws, size_t ws_size,
                              hipStream_t stream) {
  (void)in_sizes; (void)n_in; (void)out_size; (void)ws_size;
  const __hip_bfloat16* X   = (const __hip_bfloat16*)d_in[0];
  const float*          G   = (const float*)d_in[1];
  const int*            w1  = (const int*)d_in[2];
  const int*            w3  = (const int*)d_in[3];
  const int*            w2  = (const int*)d_in[4];
  const int*            s13 = (const int*)d_in[5];
  const int*            s2  = (const int*)d_in[6];
  const __hip_bfloat16* b13 = (const __hip_bfloat16*)d_in[7];
  const __hip_bfloat16* b2  = (const __hip_bfloat16*)d_in[8];
  __hip_bfloat16*       out = (__hip_bfloat16*)d_out;

  // workspace carve (requires ~34 MB)
  char* p = (char*)d_ws;
  auto carve = [&](size_t bytes) -> void* {
    void* r = (void*)p;
    p += (bytes + 255) & ~(size_t)255;
    return r;
  };
  int*            counts      = (int*)carve(E_NUM * sizeof(int));
  int*            cursor      = (int*)carve(E_NUM * sizeof(int));
  int*            base        = (int*)carve((E_NUM + 1) * sizeof(int));
  int*            top2e       = (int*)carve((size_t)T_DIM * 2 * sizeof(int));
  float*          top2w       = (float*)carve((size_t)T_DIM * 2 * sizeof(float));
  int*            slot_token  = (int*)carve((size_t)MAX_SLOTS * sizeof(int));
  float*          slot_weight = (float*)carve((size_t)MAX_SLOTS * sizeof(float));
  float*          out_accum   = (float*)carve((size_t)T_DIM * H_DIM * sizeof(float));
  __hip_bfloat16* g_ws        = (__hip_bfloat16*)carve((size_t)MAX_SLOTS * I_DIM * sizeof(__hip_bfloat16));

  init_kernel<<<dim3((T_DIM * H_DIM) / 256), 256, 0, stream>>>(counts, cursor, slot_token, slot_weight, out_accum);
  gate_kernel<<<dim3(T_DIM / 8), 256, 0, stream>>>(X, G, counts, top2e, top2w);
  scan_kernel<<<1, 1, 0, stream>>>(counts, base);
  scatter_kernel<<<dim3((T_DIM + 255) / 256), 256, 0, stream>>>(top2e, top2w, base, cursor, slot_token, slot_weight);
  ffn1_kernel<<<dim3(I_DIM / 128, MAX_SLOTS / (16 * SLOT_BLK)), 256, 0, stream>>>(X, w1, w3, s13, b13, slot_token, base, g_ws);
  ffn2_kernel<<<dim3(H_DIM / 128, MAX_SLOTS / (16 * SLOT_BLK)), 256, 0, stream>>>(g_ws, w2, s2, b2, slot_token, slot_weight, base, out_accum);
  finalize_kernel<<<dim3((T_DIM * H_DIM) / 256), 256, 0, stream>>>(out_accum, out);
}
